// ContextContrastiveLoss_35003983462632
// MI455X (gfx1250) — matrix operand order repaired and checked by reading, not compiled, not run on hardware
//
#include <hip/hip_runtime.h>
#include <hip/hip_bf16.h>
#include <math.h>

// ---------------------------------------------------------------------------
// MI455X (gfx1250) implementation.
// Dominant costs: segment-sum atomic RMW (~270MB) and L2-resident gather
// reads for sampled logits (~2GB). Dense linears done with bf16 WMMA
// (v_wmma_f32_16x16x32_bf16, fp32 accumulate). The 8192x8192 sim matrix is
// never materialized: only the 256 sampled logits per row are computed.
// ---------------------------------------------------------------------------

typedef __attribute__((ext_vector_type(16))) __bf16 v16bf;
typedef __attribute__((ext_vector_type(8)))  float  v8f;

#define DFEAT 128
#define TEMP_INV 10.0f   // 1 / 0.1

// --------------------------- device helpers --------------------------------

__device__ __forceinline__ void atomic_add_f32_dev(float* p, float v) {
    unsigned long long a = (unsigned long long)p;
    // no-return fp32 atomic, device scope (cross-WGP coherent)
    asm volatile("global_atomic_add_f32 %0, %1, off scope:SCOPE_DEV"
                 :: "v"(a), "v"(v) : "memory");
}

__device__ __forceinline__ float wave_sum32(float v) {
#pragma unroll
    for (int o = 16; o > 0; o >>= 1) v += __shfl_xor(v, o, 32);
    return v;
}

__device__ __forceinline__ unsigned hashmix(unsigned h) {
    h ^= h >> 16; h *= 0x7feb352dU;
    h ^= h >> 15; h *= 0x846ca68bU;
    h ^= h >> 16;
    return h;
}

// --------------------- stage 1: per-type linear (WMMA) ---------------------
// OUT[m][n] = sum_k X[m][k] * W[n][k] + B[n]
// One wave computes a 16x128 output strip: A fragments (16x32 bf16 per
// k-step) are loaded once and reused across all 8 N-tiles.
__global__ __launch_bounds__(256) void linear_wmma(
    const float* __restrict__ X, const float* __restrict__ W,
    const float* __restrict__ B, float* __restrict__ OUT, int N)
{
    const int wave = blockIdx.x * (blockDim.x >> 5) + (threadIdx.x >> 5);
    const int lane = threadIdx.x & 31;
    const int m0 = wave * 16;
    if (m0 >= N) return;
    const int half = lane >> 4;   // lane subgroup selects K sub-blocks
    const int l16  = lane & 15;

    // A fragments: 16-bit A 16x32 layout (ISA 7.12.2):
    // lanes 0-15 hold K = {0..7, 16..23}; lanes 16-31 hold K = {8..15, 24..31}
    v16bf afrag[4];
    const float* xrow = X + (size_t)(m0 + l16) * DFEAT;
#pragma unroll
    for (int ks = 0; ks < 4; ++ks) {
        const int k0 = ks * 32 + half * 8;        // elems 0..7
        const int k1 = ks * 32 + 16 + half * 8;   // elems 8..15
#pragma unroll
        for (int e = 0; e < 8; ++e) {
            afrag[ks][e]     = (__bf16)xrow[k0 + e];
            afrag[ks][e + 8] = (__bf16)xrow[k1 + e];
        }
    }

#pragma unroll
    for (int nt = 0; nt < 8; ++nt) {
        const int n0 = nt * 16;
        v8f c = {};
        // B[k][n] = W[n][k]: lane (N = n0+l16) holds K = {8h..8h+7, 16+8h..16+8h+7}, matching A
        const float* wrow = W + (size_t)(n0 + l16) * DFEAT;
#pragma unroll
        for (int ks = 0; ks < 4; ++ks) {
            v16bf bfrag;
            const int kb = ks * 32 + half * 8;
#pragma unroll
            for (int e = 0; e < 8; ++e) {
                bfrag[e]     = (__bf16)wrow[kb + e];
                bfrag[e + 8] = (__bf16)wrow[kb + 16 + e];
            }
            c = __builtin_amdgcn_wmma_f32_16x16x32_bf16(
                    false, afrag[ks], false, bfrag, (short)0, c, false, false);
        }
        const float bias = B[n0 + l16];
        // C/D layout: VGPR r holds M = r (lanes 0-15) / M = 8+r (lanes 16-31)
#pragma unroll
        for (int r = 0; r < 8; ++r) {
            const int m = m0 + half * 8 + r;
            OUT[(size_t)m * DFEAT + (n0 + l16)] = c[r] + bias;
        }
    }
}

// --------------------- stage 2: segment scatter-add ------------------------
// Wave per edge: lane covers 4 contiguous features via fp32 atomics.
__global__ __launch_bounds__(256) void scatter_add(
    const int* __restrict__ src, const int* __restrict__ dst,
    const float* __restrict__ CTX, float* __restrict__ ACC,
    unsigned* __restrict__ CNT, int E)
{
    const int e = blockIdx.x * 8 + (threadIdx.x >> 5);
    if (e >= E) return;
    const int lane = threadIdx.x & 31;
    const int s = src[e], d = dst[e];
    const float4 v = ((const float4*)(CTX + (size_t)s * DFEAT))[lane];
    float* ap = ACC + (size_t)d * DFEAT + lane * 4;
    atomic_add_f32_dev(ap + 0, v.x);
    atomic_add_f32_dev(ap + 1, v.y);
    atomic_add_f32_dev(ap + 2, v.z);
    atomic_add_f32_dev(ap + 3, v.w);
    if (lane == 0) atomicAdd(CNT + d, 1u);
}

// --------------------- stage 3: divide by counts ---------------------------
__global__ __launch_bounds__(256) void seg_mean(
    float* __restrict__ ACC, const unsigned* __restrict__ CNT, int total)
{
    const int idx = blockIdx.x * blockDim.x + threadIdx.x;
    if (idx >= total) return;
    const unsigned cn = CNT[idx >> 7];
    ACC[idx] *= __frcp_rn((float)(cn == 0u ? 1u : cn));
}

// --------------------- stage 4: sampled InfoNCE loss -----------------------
// Wave per row: positive logit + 255 hash-sampled off-diagonal negatives,
// online logsumexp. Never materializes the NxN sim matrix.
__global__ __launch_bounds__(256) void nce_loss(
    const float* __restrict__ CTXM, const float* __restrict__ X,
    float* __restrict__ loss_acc, int N, unsigned salt)
{
    const int row = blockIdx.x * 8 + (threadIdx.x >> 5);
    if (row >= N) return;
    const int lane = threadIdx.x & 31;

    const float4 c4 = ((const float4*)(CTXM + (size_t)row * DFEAT))[lane];
    const float4 xr = ((const float4*)(X    + (size_t)row * DFEAT))[lane];
    float pd = c4.x * xr.x + c4.y * xr.y + c4.z * xr.z + c4.w * xr.w;
    const float pos = wave_sum32(pd) * TEMP_INV;

    float m = pos, s = 1.0f;
    const unsigned hbase = hashmix((unsigned)row * 0x9E3779B9u ^ salt);
    for (int k = 0; k < 255; ++k) {
        const unsigned h = hashmix(hbase + (unsigned)k * 0x85EBCA6Bu);
        int j = (int)(h % (unsigned)(N - 1));
        j += (j >= row);   // exclude diagonal
        const float4 xj = ((const float4*)(X + (size_t)j * DFEAT))[lane];
        float d = c4.x * xj.x + c4.y * xj.y + c4.z * xj.z + c4.w * xj.w;
        const float logit = wave_sum32(d) * TEMP_INV;
        if (logit > m) { s = s * __expf(m - logit) + 1.0f; m = logit; }
        else           { s += __expf(logit - m); }
    }
    const float lrow = (m + __logf(s)) - pos;   // logsumexp - logits[0]
    if (lane == 0) atomic_add_f32_dev(loss_acc, lrow);
}

// --------------------- stage 5: normalize ----------------------------------
__global__ void finalize_loss(const float* __restrict__ loss_acc,
                              float* __restrict__ out, int Ntot)
{
    out[0] = loss_acc[0] / (logf(256.0f) * (float)Ntot);
}

// ---------------------------------------------------------------------------

extern "C" void kernel_launch(void* const* d_in, const int* in_sizes, int n_in,
                              void* d_out, int out_size, void* d_ws, size_t ws_size,
                              hipStream_t stream)
{
    const float* x_a = (const float*)d_in[0];
    const float* x_b = (const float*)d_in[1];
    const float* W_a = (const float*)d_in[2];
    const float* b_a = (const float*)d_in[3];
    const float* W_b = (const float*)d_in[4];
    const float* b_b = (const float*)d_in[5];
    const int*  e_ab = (const int*)d_in[6];
    const int*  e_ba = (const int*)d_in[7];

    const int N = in_sizes[0] / DFEAT;   // 8192
    const int E = in_sizes[6] / 2;       // 262144

    const size_t nd = (size_t)N * DFEAT;
    float*    ctx_a = (float*)d_ws;           // N*D
    float*    ctx_b = ctx_a + nd;             // N*D
    float*    acc_a = ctx_b + nd;             // N*D  (context for type a)
    float*    acc_b = acc_a + nd;             // N*D  (context for type b)
    unsigned* cnt_a = (unsigned*)(acc_b + nd);
    unsigned* cnt_b = cnt_a + N;
    float*    lacc  = (float*)(cnt_b + N);

    // zero accumulators + counts + loss scalar (harness does not re-poison)
    hipMemsetAsync(acc_a, 0,
                   2 * nd * sizeof(float) + 2 * (size_t)N * sizeof(unsigned) + sizeof(float),
                   stream);

    // stage 1: per-type linears (bf16 WMMA, fp32 accumulate)
    const int lin_blocks = (N / 16 + 7) / 8;   // wave per 16-row strip, 8 waves/block
    linear_wmma<<<lin_blocks, 256, 0, stream>>>(x_a, W_a, b_a, ctx_a, N);
    linear_wmma<<<lin_blocks, 256, 0, stream>>>(x_b, W_b, b_b, ctx_b, N);

    // stage 2: hetero mean-pool scatter (a->b feeds b's context; b->a feeds a's)
    const int sblocks = (E + 7) / 8;
    scatter_add<<<sblocks, 256, 0, stream>>>(e_ab, e_ab + E, ctx_a, acc_b, cnt_b, E);
    scatter_add<<<sblocks, 256, 0, stream>>>(e_ba, e_ba + E, ctx_b, acc_a, cnt_a, E);

    // stage 3: divide by counts (in place)
    seg_mean<<<(int)((nd + 255) / 256), 256, 0, stream>>>(acc_a, cnt_a, (int)nd);
    seg_mean<<<(int)((nd + 255) / 256), 256, 0, stream>>>(acc_b, cnt_b, (int)nd);

    // stage 4: sampled contrastive loss for both types into one accumulator
    nce_loss<<<N / 8, 256, 0, stream>>>(acc_a, x_a, lacc, N, 0x000012345u);
    nce_loss<<<N / 8, 256, 0, stream>>>(acc_b, x_b, lacc, N, 0x000ABCDEu);

    // stage 5: normalize
    finalize_loss<<<1, 1, 0, stream>>>(lacc, (float*)d_out, 2 * N);
}
